// VectorQuantizer_7438883357703
// MI455X (gfx1250) — matrix operand order repaired and checked by reading, not compiled, not run on hardware
//
#include <hip/hip_runtime.h>
#include <stddef.h>

// ---------------------------------------------------------------------------
// VQ-VAE vector quantizer for MI455X (gfx1250, wave32, WMMA).
//
// x: [8,2048,512] f32  -> N=16384 rows, D=512
// E: [8192,512]   f32  -> V=8192 codes
// argmin_col ||x-e||^2 == argmin_col ( ||e||^2 - 2*x.e )  -> 16384x512x8192 GEMM
// out = concat( qx (gathered f32 codebook rows, == qx_ste), loss )
// loss = 1.2 * mean((qx - x)^2)   (stop_gradient is value-transparent)
//
// GEMM runs in bf16 via v_wmma_f32_16x16x32_bf16 (f32 accumulate): only the
// argmin depends on the scores; gather + loss use original f32 tensors.
// B tiles stream global->LDS via GLOBAL_LOAD_ASYNC_TO_LDS_B128 (ASYNCcnt).
// ---------------------------------------------------------------------------

typedef unsigned int u32;
typedef __attribute__((ext_vector_type(16))) __bf16 v16bf;
typedef __attribute__((ext_vector_type(8)))  float  v8f;

#if defined(__HIP_DEVICE_COMPILE__) && __has_builtin(__builtin_amdgcn_global_load_async_to_lds_b128)
#define HAVE_ASYNC_LDS 1
#else
#define HAVE_ASYNC_LDS 0
#endif

#if HAVE_ASYNC_LDS
// Builtin signature (from hipcc diagnostic): first param is
//   int __attribute__((vector_size(16))) addrspace(1)*   (prints as "__device__ *")
// second is the addrspace(3) counterpart.
typedef __attribute__((vector_size(16))) int v4i_t;
typedef __attribute__((address_space(1))) v4i_t* gptr_t;
typedef __attribute__((address_space(3))) v4i_t* lptr_t;
#endif

__device__ __forceinline__ void wait_asynccnt0() {
#if defined(__HIP_DEVICE_COMPILE__)
#if __has_builtin(__builtin_amdgcn_s_wait_asynccnt)
    __builtin_amdgcn_s_wait_asynccnt(0);
#else
    asm volatile("s_wait_asynccnt 0x0" ::: "memory");
#endif
#endif
}

#define N_ROWS   16384
#define N_CODES  8192
#define DDIM     512
#define N_CHUNKS 4          // V split across 4 chunks of 2048 cols
#define ROWS_PER_WG 128     // 8 waves x 16 rows
#define JT_PER_CHUNK 128    // 2048 cols / 16

// workspace layout (bytes)
#define WS_A   ((size_t)0)                      // 16 MB  bf16 A fragments
#define WS_B   ((size_t)16777216)               //  8 MB  bf16 B fragments
#define WS_YY  ((size_t)(16777216+8388608))     // 32 KB  ||e||^2
#define WS_PV  (WS_YY + 32768)                  // 256 KB partial best vals
#define WS_PI  (WS_PV + 262144)                 // 256 KB partial best idx
#define WS_RS  (WS_PI + 262144)                 // 64 KB  per-row sum of squares

__device__ __forceinline__ u32 bf16_rne(float f) {
    u32 u = __float_as_uint(f);
    return (u + 0x7fffu + ((u >> 16) & 1u)) >> 16;   // round-to-nearest-even
}
__device__ __forceinline__ u32 bf16_pack2(float lo, float hi) {
    return bf16_rne(lo) | (bf16_rne(hi) << 16);
}

// ---------------------------------------------------------------------------
// Pack x into the CDNA5 A-fragment layout (16-bit A 16x32, ISA 7.12.2):
// lane = 16*h + M ; VGPR v holds K = (v<4?0:16) + 8h + 2*(v&3) + {0,1}
// linear index: (((rowTile*16 + kChunk)*32 + lane)*8 + v)  (one u32 = 2 bf16)
// ---------------------------------------------------------------------------
__global__ void vq_pack_a(const float* __restrict__ x, u32* __restrict__ A) {
    u32 u = blockIdx.x * 256u + threadIdx.x;      // < 4194304
    u32 v    = u & 7u;
    u32 lane = (u >> 3) & 31u;
    u32 kc   = (u >> 8) & 15u;
    u32 rt   = u >> 12;
    u32 M = lane & 15u, h = lane >> 4;
    u32 K = ((v < 4u) ? 0u : 16u) + 8u * h + 2u * (v & 3u);
    size_t base = (size_t)(rt * 16u + M) * DDIM + kc * 32u + K;
    A[u] = bf16_pack2(x[base], x[base + 1]);
}

// B-fragment layout (dual of A): lane = 16*h + N ; VGPR v holds K = (v<4?0:16) + 8h + 2*(v&3) + {0,1}
// linear index: (((colTile*16 + kChunk)*32 + lane)*8 + v)
__global__ void vq_pack_b(const float* __restrict__ E, u32* __restrict__ B) {
    u32 u = blockIdx.x * 256u + threadIdx.x;      // < 2097152
    u32 v    = u & 7u;
    u32 lane = (u >> 3) & 31u;
    u32 kc   = (u >> 8) & 15u;
    u32 j    = u >> 12;                           // col tile < 512
    u32 Nn = lane & 15u, h = lane >> 4;
    u32 K = ((v < 4u) ? 0u : 16u) + 8u * h + 2u * (v & 3u);
    size_t base = (size_t)(j * 16u + Nn) * DDIM + kc * 32u + K;
    B[u] = bf16_pack2(E[base], E[base + 1]);
}

__global__ void vq_yy(const float* __restrict__ E, float* __restrict__ yy) {
    int lane = threadIdx.x & 31, wave = threadIdx.x >> 5;
    int r = blockIdx.x * 8 + wave;                // < 8192
    const float4* e4 = (const float4*)(E + (size_t)r * DDIM);
    float s = 0.f;
#pragma unroll
    for (int i = 0; i < 4; ++i) {
        float4 q = e4[lane + i * 32];
        s += q.x * q.x + q.y * q.y + q.z * q.z + q.w * q.w;
    }
#pragma unroll
    for (int m = 1; m < 32; m <<= 1) s += __shfl_xor(s, m, 32);
    if (lane == 0) yy[r] = s;
}

// ---------------------------------------------------------------------------
// Main kernel: per wave, 16 rows x 16 cols per column tile; A resident in
// 128 VGPRs across the whole column sweep; B tile (16 KB) double-buffered in
// LDS (async global->LDS DMA when available) and shared by 8 waves; running
// per-lane argmin, reduced at the end.
// ---------------------------------------------------------------------------
__global__ void __launch_bounds__(256)
vq_gemm_argmin(const u32* __restrict__ Abuf, const u32* __restrict__ Bbuf,
               const float* __restrict__ yy,
               float* __restrict__ pval, u32* __restrict__ pidx) {
    __shared__ u32 smem[2][4096];                 // 2 x 16 KB B tiles

    const int lane  = threadIdx.x & 31;
    const int wave  = threadIdx.x >> 5;
    const int chunk = blockIdx.x;                 // 0..3
    const int rt    = blockIdx.y * 8 + wave;      // row tile 0..1023

    // ---- A fragments resident in registers: 16 chunks x v16bf (128 VGPRs)
    v16bf a[16];
    const u32* ap = Abuf + ((size_t)(rt * 16) * 32 + lane) * 8;
#pragma unroll
    for (int kc = 0; kc < 16; ++kc)
        a[kc] = *(const v16bf*)(ap + kc * 256);

    float best[8];
    u32   bidx[8];
#pragma unroll
    for (int r = 0; r < 8; ++r) { best[r] = 3.4e38f; bidx[r] = 0u; }

    const int j0 = chunk * JT_PER_CHUNK, j1 = j0 + JT_PER_CHUNK;

    // stage first B tile
#if HAVE_ASYNC_LDS
    {
        gptr_t gp = (gptr_t)(Bbuf + (size_t)j0 * 4096 + threadIdx.x * 4);
        lptr_t lp = (lptr_t)&smem[j0 & 1][threadIdx.x * 4];
        __builtin_amdgcn_global_load_async_to_lds_b128(gp, lp, 0, 0);
        __builtin_amdgcn_global_load_async_to_lds_b128(gp, lp, 4096, 0);
        __builtin_amdgcn_global_load_async_to_lds_b128(gp, lp, 8192, 0);
        __builtin_amdgcn_global_load_async_to_lds_b128(gp, lp, 12288, 0);
    }
    wait_asynccnt0();
#else
    {
        const uint4* gb = (const uint4*)(Bbuf + (size_t)j0 * 4096);
        uint4* sb = (uint4*)smem[j0 & 1];
#pragma unroll
        for (int i = 0; i < 4; ++i) sb[threadIdx.x + i * 256] = gb[threadIdx.x + i * 256];
    }
#endif
    __syncthreads();

    for (int j = j0; j < j1; ++j) {
        const int cur = j & 1, nxt = cur ^ 1;
        const bool more = (j + 1 < j1);
        // prefetch tile j+2 into L2 (global_prefetch_b8)
        __builtin_prefetch(Bbuf + (size_t)(j + 2) * 4096, 0, 1);

#if HAVE_ASYNC_LDS
        // DMA next tile straight into LDS; overlaps with the WMMAs below.
        if (more) {
            gptr_t gp = (gptr_t)(Bbuf + (size_t)(j + 1) * 4096 + threadIdx.x * 4);
            lptr_t lp = (lptr_t)&smem[nxt][threadIdx.x * 4];
            __builtin_amdgcn_global_load_async_to_lds_b128(gp, lp, 0, 0);
            __builtin_amdgcn_global_load_async_to_lds_b128(gp, lp, 4096, 0);
            __builtin_amdgcn_global_load_async_to_lds_b128(gp, lp, 8192, 0);
            __builtin_amdgcn_global_load_async_to_lds_b128(gp, lp, 12288, 0);
        }
#else
        // fallback: stage through VGPRs (loads now, LDS stores after WMMAs)
        uint4 n0, n1, n2, n3;
        if (more) {
            const uint4* gb = (const uint4*)(Bbuf + (size_t)(j + 1) * 4096);
            n0 = gb[threadIdx.x];       n1 = gb[threadIdx.x + 256];
            n2 = gb[threadIdx.x + 512]; n3 = gb[threadIdx.x + 768];
        }
#endif

        // ---- K loop: 16 x v_wmma_f32_16x16x32_bf16, 2 accumulators for ILP
        v8f acc0 = {}, acc1 = {};
        const u32* bl = &smem[cur][lane * 8];
#pragma unroll
        for (int kc = 0; kc < 16; kc += 2) {
            v16bf b0 = *(const v16bf*)(bl + kc * 256);
            v16bf b1 = *(const v16bf*)(bl + (kc + 1) * 256);
            acc0 = __builtin_amdgcn_wmma_f32_16x16x32_bf16(
                false, a[kc],     false, b0, (short)0, acc0, false, false);
            acc1 = __builtin_amdgcn_wmma_f32_16x16x32_bf16(
                false, a[kc + 1], false, b1, (short)0, acc1, false, false);
        }

#if !HAVE_ASYNC_LDS
        if (more) {
            uint4* sb = (uint4*)smem[nxt];
            sb[threadIdx.x] = n0;       sb[threadIdx.x + 256] = n1;
            sb[threadIdx.x + 512] = n2; sb[threadIdx.x + 768] = n3;
        }
#endif

        // ---- epilogue: d = ||e||^2 - 2*score ; running per-lane argmin
        const u32 col = (u32)(j * 16 + (lane & 15));
        const float yv = yy[col];
#pragma unroll
        for (int r = 0; r < 8; ++r) {
            float d = yv - 2.0f * (acc0[r] + acc1[r]);
            if (d < best[r]) { best[r] = d; bidx[r] = col; }
        }

#if HAVE_ASYNC_LDS
        if (more) wait_asynccnt0();   // my DMA writes to smem[nxt] done
#endif
        __syncthreads();              // all waves done reading/writing tiles
    }

    // ---- cross-lane reduce within each 16-lane half (C layout: VGPR r ->
    // rows r (lanes 0-15) and r+8 (lanes 16-31)); tie-break on smaller index.
#pragma unroll
    for (int r = 0; r < 8; ++r) {
        float v = best[r];
        u32   i = bidx[r];
#pragma unroll
        for (int m = 1; m < 16; m <<= 1) {
            float ov = __shfl_xor(v, m, 32);
            u32   oi = (u32)__shfl_xor((int)i, m, 32);
            if (ov < v || (ov == v && oi < i)) { v = ov; i = oi; }
        }
        if ((lane & 15) == 0) {
            int row = rt * 16 + r + (lane >> 4) * 8;
            pval[chunk * N_ROWS + row] = v;
            pidx[chunk * N_ROWS + row] = i;
        }
    }
}

// ---------------------------------------------------------------------------
// Gather q = E[code] (f32), write qx_ste (= qx), per-row sum of (q-x)^2.
// ---------------------------------------------------------------------------
__global__ void vq_gather(const float* __restrict__ x, const float* __restrict__ E,
                          const float* __restrict__ pv, const u32* __restrict__ pi,
                          float* __restrict__ out, float* __restrict__ rowsum) {
    int lane = threadIdx.x & 31, wave = threadIdx.x >> 5;
    int row = blockIdx.x * 8 + wave;              // < 16384
    float bv = pv[row];
    u32   bi = pi[row];
#pragma unroll
    for (int c = 1; c < N_CHUNKS; ++c) {
        float v2 = pv[c * N_ROWS + row];
        u32   i2 = pi[c * N_ROWS + row];
        if (v2 < bv || (v2 == bv && i2 < bi)) { bv = v2; bi = i2; }
    }
    const float4* e4 = (const float4*)(E + (size_t)bi * DDIM);
    const float4* x4 = (const float4*)(x + (size_t)row * DDIM);
    float4* o4 = (float4*)(out + (size_t)row * DDIM);
    float s = 0.f;
#pragma unroll
    for (int i = 0; i < 4; ++i) {
        float4 q = e4[lane + i * 32], xv = x4[lane + i * 32];
        o4[lane + i * 32] = q;
        float dx = q.x - xv.x, dy = q.y - xv.y, dz = q.z - xv.z, dw = q.w - xv.w;
        s += dx * dx + dy * dy + dz * dz + dw * dw;
    }
#pragma unroll
    for (int m = 1; m < 32; m <<= 1) s += __shfl_xor(s, m, 32);
    if (lane == 0) rowsum[row] = s;
}

// Deterministic final reduction (fixed order, no float atomics).
__global__ void vq_loss(const float* __restrict__ rowsum, float* __restrict__ out_loss) {
    __shared__ float sm[256];
    float s = 0.f;
    for (int i = threadIdx.x; i < N_ROWS; i += 256) s += rowsum[i];
    sm[threadIdx.x] = s;
    __syncthreads();
    for (int off = 128; off > 0; off >>= 1) {
        if ((int)threadIdx.x < off) sm[threadIdx.x] += sm[threadIdx.x + off];
        __syncthreads();
    }
    if (threadIdx.x == 0)
        *out_loss = 1.2f * sm[0] / (float)(N_ROWS * DDIM);   // (1 + BETA) * mean
}

extern "C" void kernel_launch(void* const* d_in, const int* in_sizes, int n_in,
                              void* d_out, int out_size, void* d_ws, size_t ws_size,
                              hipStream_t stream) {
    const float* x = (const float*)d_in[0];       // 8*2048*512
    const float* E = (const float*)d_in[1];       // 8192*512
    float* out = (float*)d_out;                   // qx (8388608) + loss (1)
    char* ws = (char*)d_ws;

    u32*   wsA = (u32*)(ws + WS_A);
    u32*   wsB = (u32*)(ws + WS_B);
    float* yy  = (float*)(ws + WS_YY);
    float* pv  = (float*)(ws + WS_PV);
    u32*   pi  = (u32*)(ws + WS_PI);
    float* rs  = (float*)(ws + WS_RS);

    vq_pack_a<<<16384, 256, 0, stream>>>(x, wsA);
    vq_pack_b<<<8192, 256, 0, stream>>>(E, wsB);
    vq_yy<<<N_CODES / 8, 256, 0, stream>>>(E, yy);
    vq_gemm_argmin<<<dim3(N_CHUNKS, N_ROWS / ROWS_PER_WG), 256, 0, stream>>>(
        wsA, wsB, yy, pv, pi);
    vq_gather<<<N_ROWS / 8, 256, 0, stream>>>(x, E, pv, pi, out, rs);
    vq_loss<<<1, 256, 0, stream>>>(rs, out + (size_t)N_ROWS * DDIM);
}
